// PositionAttention_17789754540756
// MI455X (gfx1250) — matrix operand order repaired and checked by reading, not compiled, not run on hardware
//
#include <hip/hip_runtime.h>

// ---- CDNA5 (gfx1250, wave32) types -----------------------------------------
typedef __attribute__((ext_vector_type(16))) __bf16 v16bf;
typedef __attribute__((ext_vector_type(8)))  __bf16 v8bf;
typedef __attribute__((ext_vector_type(8)))  float  v8f;

// Problem constants (from the reference: b=4, c=124, h=w=64)
constexpr int B_ = 4;
constexpr int C_ = 124;
constexpr int D_ = 128;    // c padded to 128 for WMMA K-dim
constexpr int N_ = 4096;   // h*w tokens

__device__ __forceinline__ v16bf ld16(const __bf16* lo, const __bf16* hi) {
    union { v16bf v; v8bf h[2]; } u;
    u.h[0] = *(const v8bf*)lo;
    u.h[1] = *(const v8bf*)hi;
    return u.v;
}

// Butterfly max over the 16-lane half-group using DPP16 ROW_XMASK (no LDS).
__device__ __forceinline__ float rowmax16(float v) {
    float t;
    t = __int_as_float(__builtin_amdgcn_update_dpp(0, __float_as_int(v), 0x161, 0xf, 0xf, true));
    v = fmaxf(v, t);
    t = __int_as_float(__builtin_amdgcn_update_dpp(0, __float_as_int(v), 0x162, 0xf, 0xf, true));
    v = fmaxf(v, t);
    t = __int_as_float(__builtin_amdgcn_update_dpp(0, __float_as_int(v), 0x164, 0xf, 0xf, true));
    v = fmaxf(v, t);
    t = __int_as_float(__builtin_amdgcn_update_dpp(0, __float_as_int(v), 0x168, 0xf, 0xf, true));
    v = fmaxf(v, t);
    return v;
}

// ---- Phase 0: Bf = Wb @ x, emitted in two bf16 layouts ---------------------
//   BfT [b][n][128] token-major  (Q rows / K rows; channels 124..127 = 0)
//   Bfc [b][128][n] channel-major (V columns; channels 124..127 = 0)
__global__ __launch_bounds__(256) void conv1x1_bf16_kernel(
    const float* __restrict__ x, const float* __restrict__ Wb,
    __bf16* __restrict__ BfT, __bf16* __restrict__ Bfc)
{
    const int t = blockIdx.x * blockDim.x + threadIdx.x;  // token, coalesced
    const int o = blockIdx.y;                             // out channel 0..127
    const int b = blockIdx.z;
    float acc = 0.f;
    if (o < C_) {
        const float* xr = x + (size_t)b * C_ * N_ + t;
        const float* wr = Wb + (size_t)o * C_;            // uniform -> SMEM loads
        #pragma unroll 4
        for (int c = 0; c < C_; ++c)
            acc = fmaf(wr[c], xr[(size_t)c * N_], acc);
    }
    const __bf16 v = (__bf16)acc;
    Bfc[((size_t)b * D_ + o) * N_ + t] = v;
    BfT[((size_t)b * N_ + t) * D_ + o] = v;
}

// ---- Phase 1: flash attention, Q=K=V=Bf, out = gamma*E + x -----------------
// 4 waves/WG, each wave owns a 16-row m-tile; 64-column K/V chunks.
__global__ __launch_bounds__(128) void pa_flash_kernel(
    const float* __restrict__ x, const float* __restrict__ gamma,
    const __bf16* __restrict__ BfT, const __bf16* __restrict__ Bfc,
    float* __restrict__ out)
{
    __shared__ __align__(16) __bf16 ps[4][16][64];   // per-wave P staging (C->A layout)

    const int b    = blockIdx.y;
    const int wave = threadIdx.x >> 5;
    const int lane = threadIdx.x & 31;
    const int col  = lane & 15;    // A: M row / B: N col / C: N col
    const int hi   = lane >> 4;    // A: K group / B: K half / C: row-half
    const int m0   = blockIdx.x * 64 + wave * 16;

    // ones B-operand for WMMA row-sum (sum_k P[m][k] replicated over N cols)
    v16bf ones;
    #pragma unroll
    for (int i = 0; i < 16; ++i) ones[i] = (__bf16)1.0f;
    const v8f vzero = {};

    // Q A-operand tiles: row m0+col, K groups at dc*32 + {hi*8, 16+hi*8}
    const __bf16* q = BfT + ((size_t)b * N_ + m0 + col) * D_;
    v16bf qa[4];
    #pragma unroll
    for (int dc = 0; dc < 4; ++dc)
        qa[dc] = ld16(q + dc * 32 + hi * 8, q + dc * 32 + 16 + hi * 8);

    v8f o8[8] = {};                 // E^T accumulators: 16 rows x 128 cols
    v8f rs = {};                    // running row sums (replicated per half)
    float rm[8];                    // running row max
    #pragma unroll
    for (int r = 0; r < 8; ++r) rm[r] = -3.0e38f;

    const __bf16* kB = BfT + (size_t)b * N_ * D_;
    const __bf16* vB = Bfc + (size_t)b * D_ * N_;

    for (int j = 0; j < N_; j += 64) {
        // --- scores: four 16x16 tiles over K=128 (4 WMMAs each) ---
        v8f s[4] = {};
        #pragma unroll
        for (int dc = 0; dc < 4; ++dc) {
            #pragma unroll
            for (int t = 0; t < 4; ++t) {
                const __bf16* kt = kB + (size_t)(j + 16 * t + col) * D_ + dc * 32 + hi * 8;
                v16bf bt = ld16(kt, kt + 16);
                s[t] = __builtin_amdgcn_wmma_f32_16x16x32_bf16(false, qa[dc], false, bt,
                                                               (short)0, s[t], false, false);
            }
        }

        // --- online softmax over the 64-column chunk ---
        float nm[8];
        int changed = 0;
        #pragma unroll
        for (int r = 0; r < 8; ++r) {
            float cm = fmaxf(fmaxf(s[0][r], s[1][r]), fmaxf(s[2][r], s[3][r]));
            cm = rowmax16(cm);
            nm[r] = fmaxf(rm[r], cm);
            changed |= (nm[r] > rm[r]) ? 1 : 0;
        }
        if (__any(changed)) {        // wave-uniform; rare after warm-up
            v8f scv;
            #pragma unroll
            for (int r = 0; r < 8; ++r) scv[r] = __expf(rm[r] - nm[r]);
            rs = rs * scv;
            #pragma unroll
            for (int dt = 0; dt < 8; ++dt) o8[dt] = o8[dt] * scv;
        }
        #pragma unroll
        for (int r = 0; r < 8; ++r) rm[r] = nm[r];

        // exp + stage P into LDS (C-layout -> row-major [m][k] tile)
        #pragma unroll
        for (int r = 0; r < 8; ++r) {
            const int m = r + 8 * hi;
            #pragma unroll
            for (int t = 0; t < 4; ++t) {
                const float p = __expf(s[t][r] - rm[r]);
                ps[wave][m][col + 16 * t] = (__bf16)p;
            }
        }

        asm volatile("s_wait_dscnt 0" ::: "memory");     // wave-private LDS, in-order
        const __bf16* pr = &ps[wave][col][0];
        v16bf pa0 = ld16(pr + hi * 8,      pr + 16 + hi * 8);
        v16bf pa1 = ld16(pr + 32 + hi * 8, pr + 48 + hi * 8);

        // row-sum of P via WMMA against ones (replicated across cols in D tile)
        v8f psum = __builtin_amdgcn_wmma_f32_16x16x32_bf16(false, pa0, false, ones,
                                                           (short)0, vzero, false, false);
        psum = __builtin_amdgcn_wmma_f32_16x16x32_bf16(false, pa1, false, ones,
                                                       (short)0, psum, false, false);

        // --- O += P(16x64) x V(64x16) for each of 8 d-tiles ---
        #pragma unroll
        for (int dt = 0; dt < 8; ++dt) {
            const __bf16* vr = vB + (size_t)(dt * 16 + col) * N_ + j + hi * 8;
            v16bf vb0 = ld16(vr, vr + 16);
            v16bf vb1 = ld16(vr + 32, vr + 48);
            o8[dt] = __builtin_amdgcn_wmma_f32_16x16x32_bf16(false, pa0, false, vb0,
                                                             (short)0, o8[dt], false, false);
            o8[dt] = __builtin_amdgcn_wmma_f32_16x16x32_bf16(false, pa1, false, vb1,
                                                             (short)0, o8[dt], false, false);
        }

        rs = rs + psum;
    }

    // --- normalize, scale by gamma, add residual x ---
    const float g = gamma[0];
    float inv[8];
    #pragma unroll
    for (int r = 0; r < 8; ++r) inv[r] = 1.0f / rs[r];
    #pragma unroll
    for (int dt = 0; dt < 8; ++dt) {
        const int c = dt * 16 + col;
        if (c < C_) {
            #pragma unroll
            for (int r = 0; r < 8; ++r) {
                const int m = m0 + r + 8 * hi;
                const size_t idx = ((size_t)(b * C_ + c)) * N_ + m;
                out[idx] = g * o8[dt][r] * inv[r] + x[idx];
            }
        }
    }
}

// ---- Host launcher ---------------------------------------------------------
extern "C" void kernel_launch(void* const* d_in, const int* in_sizes, int n_in,
                              void* d_out, int out_size, void* d_ws, size_t ws_size,
                              hipStream_t stream) {
    (void)in_sizes; (void)n_in; (void)out_size; (void)ws_size;
    const float* x     = (const float*)d_in[0];
    const float* Wb    = (const float*)d_in[1];
    const float* gamma = (const float*)d_in[2];
    float* out = (float*)d_out;

    __bf16* BfT = (__bf16*)d_ws;                       // 4*4096*128*2 = 4 MB
    __bf16* Bfc = BfT + (size_t)B_ * N_ * D_;          // 4 MB

    dim3 g0(N_ / 256, D_, B_);
    conv1x1_bf16_kernel<<<g0, 256, 0, stream>>>(x, Wb, BfT, Bfc);

    dim3 g1(N_ / 64, B_);
    pa_flash_kernel<<<g1, 128, 0, stream>>>(x, gamma, BfT, Bfc, out);
}
